// LSTM_24232205484460
// MI455X (gfx1250) — matrix operand order repaired and checked by reading, not compiled, not run on hardware
//
#include <hip/hip_runtime.h>
#include <stdint.h>
#include <math.h>

#define H_DIM  2048
#define X_DIM  2048
#define T_LEN  512
#define N4     8192   // 4 * H_DIM (gates i,f,c,o stacked along N)
#define REC_BLOCKS 128

typedef __attribute__((ext_vector_type(16))) __bf16 v16bf;
typedef __attribute__((ext_vector_type(8)))  float  v8f;

// ---------- fp32 <-> bf16 (RNE) helpers on raw bits ----------
__device__ __forceinline__ unsigned short f32_to_bf16_rne(float f) {
    unsigned u = __float_as_uint(f);
    u += 0x7fffu + ((u >> 16) & 1u);
    return (unsigned short)(u >> 16);
}
__device__ __forceinline__ float bf16us_to_f32(unsigned short s) {
    return __uint_as_float(((unsigned)s) << 16);
}

// =====================================================================
// Kernel 1: init state + grid barrier
// =====================================================================
__global__ void init_state(float* __restrict__ h0, float* __restrict__ c0,
                           unsigned* __restrict__ bar) {
    int i = blockIdx.x * blockDim.x + threadIdx.x;
    if (i < H_DIM) { h0[i] = 0.0f; c0[i] = 0.0f; }
    if (i == 0) bar[0] = 0u;
}

// =====================================================================
// Kernel 2: fused input-projection GEMM, split-bf16 WMMA, f32 accum.
// P[t][n] = sum_k xs[t][k] * Wg[n%2048][k] + bias_g[n%2048],  n -> gate g = n/2048
// Block: 256 thr = 8 waves, tile M=32 x N=64 (waves 2x4, 16x16 each).
// =====================================================================
__global__ __launch_bounds__(256)
void proj_gemm(const float* __restrict__ xs,
               const float* __restrict__ Wix, const float* __restrict__ Wfx,
               const float* __restrict__ Wcx, const float* __restrict__ Wox,
               const float* __restrict__ bi,  const float* __restrict__ bf_,
               const float* __restrict__ bc,  const float* __restrict__ bo,
               float* __restrict__ P)
{
    __shared__ unsigned short Ah[32][36], Al[32][36];   // xs tile  (M x K) hi/lo
    __shared__ unsigned short Bh[64][36], Bl[64][36];   // W  tile  (N x K) hi/lo

    const int tid   = threadIdx.x;
    const int lane  = tid & 31;
    const int wid   = tid >> 5;
    const int waveM = wid >> 2;          // 0..1
    const int waveN = wid & 3;           // 0..3
    const int half  = lane >> 4;         // 0: lanes 0-15, 1: lanes 16-31
    const int lrow  = lane & 15;

    const int nBlock = blockIdx.x * 64;            // global N base (0..8191)
    const int mBlock = blockIdx.y * 32;            // global M base (time)
    const int gate   = nBlock >> 11;               // 0..3
    const int wBase  = nBlock & (H_DIM - 1);       // row base inside gate

    const float* __restrict__ Wg =
        (gate == 0) ? Wix : (gate == 1) ? Wfx : (gate == 2) ? Wcx : Wox;
    const float* __restrict__ bg =
        (gate == 0) ? bi  : (gate == 1) ? bf_ : (gate == 2) ? bc  : bo;

    v8f acc = {};

    const int arow = tid >> 3, acol = (tid & 7) << 2;   // A coop load: 4 f32/thr
    const int brow = tid >> 2, bcol = (tid & 3) << 3;   // B coop load: 8 f32/thr

    for (int k0 = 0; k0 < X_DIM; k0 += 32) {
        // ---- stage A tile (32x32 fp32 -> hi/lo bf16) ----
        {
            const float4 v = *(const float4*)(xs + (size_t)(mBlock + arow) * X_DIM + k0 + acol);
            float a[4] = {v.x, v.y, v.z, v.w};
            #pragma unroll
            for (int q = 0; q < 4; ++q) {
                unsigned short hi = f32_to_bf16_rne(a[q]);
                Ah[arow][acol + q] = hi;
                Al[arow][acol + q] = f32_to_bf16_rne(a[q] - bf16us_to_f32(hi));
            }
        }
        // ---- stage B tile (64x32 fp32 -> hi/lo bf16) ----
        {
            const float* src = Wg + (size_t)(wBase + brow) * X_DIM + k0 + bcol;
            #pragma unroll
            for (int h2 = 0; h2 < 2; ++h2) {
                const float4 v = *(const float4*)(src + h2 * 4);
                float a[4] = {v.x, v.y, v.z, v.w};
                #pragma unroll
                for (int q = 0; q < 4; ++q) {
                    unsigned short hi = f32_to_bf16_rne(a[q]);
                    Bh[brow][bcol + h2 * 4 + q] = hi;
                    Bl[brow][bcol + h2 * 4 + q] = f32_to_bf16_rne(a[q] - bf16us_to_f32(hi));
                }
            }
        }
        __syncthreads();

        // ---- build fragments per ISA 16-bit layouts ----
        v16bf a_hi, a_lo, b_hi, b_lo;
        const unsigned short* arh = &Ah[waveM * 16 + lrow][0];
        const unsigned short* arl = &Al[waveM * 16 + lrow][0];
        const unsigned short* brh = &Bh[waveN * 16 + lrow][0];
        const unsigned short* brl = &Bl[waveN * 16 + lrow][0];
        #pragma unroll
        for (int e = 0; e < 16; ++e) {
            const int ka = (e & 7) + ((e >> 3) << 4) + (half << 3); // A: 16x32 layout
            const int kb = (e & 7) + ((e >> 3) << 4) + (half << 3); // B: 32x16 layout (same k order as A)
            a_hi[e] = __builtin_bit_cast(__bf16, arh[ka]);
            a_lo[e] = __builtin_bit_cast(__bf16, arl[ka]);
            b_hi[e] = __builtin_bit_cast(__bf16, brh[kb]);
            b_lo[e] = __builtin_bit_cast(__bf16, brl[kb]);
        }
        // split-bf16 compensated product: hi*hi + hi*lo + lo*hi  (~fp32 accurate)
        acc = __builtin_amdgcn_wmma_f32_16x16x32_bf16(false, a_hi, false, b_hi, (short)0, acc, false, false);
        acc = __builtin_amdgcn_wmma_f32_16x16x32_bf16(false, a_hi, false, b_lo, (short)0, acc, false, false);
        acc = __builtin_amdgcn_wmma_f32_16x16x32_bf16(false, a_lo, false, b_hi, (short)0, acc, false, false);
        __syncthreads();
    }

    // ---- epilogue: C/D layout -> P, add bias ----
    const int colg  = nBlock + waveN * 16 + lrow;          // 0..8191
    const float bias = bg[wBase + waveN * 16 + lrow];
    const int rbase = mBlock + waveM * 16 + half * 8;
    #pragma unroll
    for (int j = 0; j < 8; ++j)
        P[(size_t)(rbase + j) * N4 + colg] = acc[j] + bias;
}

// =====================================================================
// Grid-wide barrier (ticket / generation). Reset to 0 each call by init.
// =====================================================================
__device__ __forceinline__ void grid_sync(unsigned* bar, unsigned nblk) {
    __syncthreads();
    if (threadIdx.x == 0) {
        __threadfence();
        unsigned ticket = __hip_atomic_fetch_add(bar, 1u, __ATOMIC_ACQ_REL,
                                                 __HIP_MEMORY_SCOPE_AGENT);
        unsigned target = (ticket / nblk + 1u) * nblk;
        while (__hip_atomic_load(bar, __ATOMIC_ACQUIRE,
                                 __HIP_MEMORY_SCOPE_AGENT) < target) {
            __builtin_amdgcn_s_sleep(2);
        }
        __threadfence();
    }
    __syncthreads();
}

// =====================================================================
// Kernel 3: persistent recurrence. 128 blocks x 256 thr = 1024 waves.
// Recurrent weights (112 MB fp32) stream from the 192 MB L2 each step.
// Phase A: i,f,c~ matvecs -> c_new ; sync ; Phase B: o matvec -> h_new ; sync.
// =====================================================================
__global__ __launch_bounds__(256)
void lstm_recurrence(const float* __restrict__ P,
                     const float* __restrict__ Wih, const float* __restrict__ Wic,
                     const float* __restrict__ Wfh, const float* __restrict__ Wfc,
                     const float* __restrict__ Wch,
                     const float* __restrict__ Woh, const float* __restrict__ Woc,
                     float* __restrict__ h0, float* __restrict__ h1,
                     float* __restrict__ c0, float* __restrict__ c1,
                     unsigned* __restrict__ bar,
                     float* __restrict__ out)
{
    __shared__ float sh[H_DIM];   // h_prev (block-local copy)
    __shared__ float sc[H_DIM];   // c_prev, then c_new in phase B

    const int tid   = threadIdx.x;
    const int lane  = tid & 31;
    const int wid   = tid >> 5;
    const unsigned nblk = gridDim.x;
    const int gwave = blockIdx.x * 8 + wid;
    const int nwave = (int)nblk * 8;

    for (int t = 0; t < T_LEN; ++t) {
        const float* hin   = (t & 1) ? h1 : h0;
        float*       hout  = (t & 1) ? h0 : h1;
        const float* cin   = (t & 1) ? c1 : c0;
        float*       cnew  = (t & 1) ? c0 : c1;
        const float* Pt    = P + (size_t)t * N4;

        for (int i = tid; i < H_DIM; i += 256) { sh[i] = hin[i]; sc[i] = cin[i]; }
        __syncthreads();

        // ---- Phase A: gates i, f, c~  ->  c_new ----
        for (int r = gwave; r < H_DIM; r += nwave) {
            const float* wih = Wih + (size_t)r * H_DIM;
            const float* wic = Wic + (size_t)r * H_DIM;
            const float* wfh = Wfh + (size_t)r * H_DIM;
            const float* wfc = Wfc + (size_t)r * H_DIM;
            const float* wch = Wch + (size_t)r * H_DIM;
            float ai = 0.f, af = 0.f, ac = 0.f;
            for (int k = lane * 4; k < H_DIM; k += 128) {
                const float4 hv = *(const float4*)(sh + k);
                const float4 cv = *(const float4*)(sc + k);
                const float4 w1 = *(const float4*)(wih + k);
                const float4 w2 = *(const float4*)(wic + k);
                const float4 w3 = *(const float4*)(wfh + k);
                const float4 w4 = *(const float4*)(wfc + k);
                const float4 w5 = *(const float4*)(wch + k);
                ai += w1.x*hv.x + w1.y*hv.y + w1.z*hv.z + w1.w*hv.w
                    + w2.x*cv.x + w2.y*cv.y + w2.z*cv.z + w2.w*cv.w;
                af += w3.x*hv.x + w3.y*hv.y + w3.z*hv.z + w3.w*hv.w
                    + w4.x*cv.x + w4.y*cv.y + w4.z*cv.z + w4.w*cv.w;
                ac += w5.x*hv.x + w5.y*hv.y + w5.z*hv.z + w5.w*hv.w;
            }
            #pragma unroll
            for (int off = 16; off > 0; off >>= 1) {
                ai += __shfl_down(ai, off, 32);
                af += __shfl_down(af, off, 32);
                ac += __shfl_down(ac, off, 32);
            }
            if (lane == 0) {
                const float ig = 1.f / (1.f + expf(-(Pt[r] + ai)));
                const float fg = 1.f / (1.f + expf(-(Pt[H_DIM + r] + af)));
                const float cc = tanhf(Pt[2 * H_DIM + r] + ac);
                cnew[r] = fg * sc[r] + ig * cc;
            }
        }
        grid_sync(bar, nblk);   // c_new complete device-wide

        for (int i = tid; i < H_DIM; i += 256) sc[i] = cnew[i];
        __syncthreads();

        // ---- Phase B: gate o (peephole on NEW cell) -> h_new ----
        for (int r = gwave; r < H_DIM; r += nwave) {
            const float* woh = Woh + (size_t)r * H_DIM;
            const float* woc = Woc + (size_t)r * H_DIM;
            float ao = 0.f;
            for (int k = lane * 4; k < H_DIM; k += 128) {
                const float4 hv = *(const float4*)(sh + k);
                const float4 cv = *(const float4*)(sc + k);
                const float4 w1 = *(const float4*)(woh + k);
                const float4 w2 = *(const float4*)(woc + k);
                ao += w1.x*hv.x + w1.y*hv.y + w1.z*hv.z + w1.w*hv.w
                    + w2.x*cv.x + w2.y*cv.y + w2.z*cv.z + w2.w*cv.w;
            }
            #pragma unroll
            for (int off = 16; off > 0; off >>= 1)
                ao += __shfl_down(ao, off, 32);
            if (lane == 0) {
                const float og = 1.f / (1.f + expf(-(Pt[3 * H_DIM + r] + ao)));
                const float hn = og * tanhf(sc[r]);
                hout[r] = hn;
                if (t == T_LEN - 1) out[r] = hn;
            }
        }
        grid_sync(bar, nblk);   // h_new complete before next step
    }
}

// =====================================================================
extern "C" void kernel_launch(void* const* d_in, const int* in_sizes, int n_in,
                              void* d_out, int out_size, void* d_ws, size_t ws_size,
                              hipStream_t stream) {
    (void)in_sizes; (void)n_in; (void)out_size; (void)ws_size;

    const float* xs  = (const float*)d_in[0];
    const float* Wix = (const float*)d_in[1];
    const float* Wih = (const float*)d_in[2];
    const float* Wic = (const float*)d_in[3];
    const float* Wfx = (const float*)d_in[4];
    const float* Wfh = (const float*)d_in[5];
    const float* Wfc = (const float*)d_in[6];
    const float* Wcx = (const float*)d_in[7];
    const float* Wch = (const float*)d_in[8];
    const float* Wox = (const float*)d_in[9];
    const float* Woh = (const float*)d_in[10];
    const float* Woc = (const float*)d_in[11];
    const float* bi  = (const float*)d_in[12];
    const float* bff = (const float*)d_in[13];
    const float* bc  = (const float*)d_in[14];
    const float* bo  = (const float*)d_in[15];

    float* P  = (float*)d_ws;                       // 512*8192 f32 = 16 MB
    float* h0 = P + (size_t)T_LEN * N4;
    float* h1 = h0 + H_DIM;
    float* c0 = h1 + H_DIM;
    float* c1 = c0 + H_DIM;
    unsigned* bar = (unsigned*)(c1 + H_DIM);

    init_state<<<8, 256, 0, stream>>>(h0, c0, bar);

    dim3 ggrid(N4 / 64, T_LEN / 32);                // 128 x 16 blocks
    proj_gemm<<<ggrid, 256, 0, stream>>>(xs, Wix, Wfx, Wcx, Wox,
                                         bi, bff, bc, bo, P);

    lstm_recurrence<<<REC_BLOCKS, 256, 0, stream>>>(P, Wih, Wic, Wfh, Wfc, Wch,
                                                    Woh, Woc, h0, h1, c0, c1,
                                                    bar, (float*)d_out);
}
